// InteractionNetwork_37787122270588
// MI455X (gfx1250) — matrix operand order repaired and checked by reading, not compiled, not run on hardware
//
#include <hip/hip_runtime.h>
#include <hip/hip_fp16.h>

typedef __attribute__((ext_vector_type(16))) _Float16 v16h;
typedef __attribute__((ext_vector_type(8)))  float    v8f;

#define DEV static __device__ __forceinline__

constexpr int   kN   = 50000;
constexpr int   kE   = 1600000;
constexpr int   kG   = 256;
constexpr float kEPS = 1e-5f;
constexpr int   TPB  = 8;   // 16-row tiles per block

// ---------------------------------------------------------------- async staging
#if defined(__HIP_DEVICE_COMPILE__)
#if __has_builtin(__builtin_amdgcn_global_load_async_to_lds_b128) && \
    __has_builtin(__builtin_amdgcn_s_wait_asynccnt)
#define HAVE_ASYNC_LDS 1
#endif
#endif
#ifndef HAVE_ASYNC_LDS
#define HAVE_ASYNC_LDS 0
#endif

#if HAVE_ASYNC_LDS
typedef int v4i __attribute__((vector_size(16)));
typedef __attribute__((address_space(1))) v4i v4i_g;  // global
typedef __attribute__((address_space(3))) v4i v4i_l;  // LDS
#endif

// issue one 16B global->LDS copy per thread (async path when available)
DEV void stage16B_issue(const _Float16* g, _Float16* l, int t) {
#if HAVE_ASYNC_LDS
  __builtin_amdgcn_global_load_async_to_lds_b128(
      (v4i_g*)(g + t * 8), (v4i_l*)(l + t * 8), 0, 0);
#else
  ((uint4*)l)[t] = ((const uint4*)g)[t];
#endif
}
DEV void stage_wait() {
#if HAVE_ASYNC_LDS
  __builtin_amdgcn_s_wait_asynccnt(0);
#endif
}

// ---------------------------------------------------------------- WMMA helpers
DEV v8f wmma16(v16h a, v16h b, v8f c) {
  return __builtin_amdgcn_wmma_f32_16x16x32_f16(false, a, false, b, (short)0, c,
                                                false, false);
}

// A fragment (16x32 f16, MxK) from LDS row-major [16][lda]
DEV v16h frag_a(const _Float16* s, int lda, int k0, int lane) {
  const int g = lane >> 4, m = lane & 15;
  v16h a;
#pragma unroll
  for (int i = 0; i < 16; ++i) {
    const int v = i >> 1, p = i & 1;
    const int k = k0 + 2 * v + ((v >> 2) << 3) + p + (g << 3);
    a[i] = s[m * lda + k];
  }
  return a;
}

// B fragment (32x16 f16, KxN) from global weights row-major [K][128], zero pad k>=K
DEV v16h frag_b(const _Float16* w, int K, int n0, int k0, int lane) {
  const int g = lane >> 4, n = (lane & 15) + n0;
  v16h b;
#pragma unroll
  for (int i = 0; i < 16; ++i) {
    const int k = k0 + (i & 7) + ((i >> 3) << 4) + (g << 3);
    b[i] = (k < K) ? w[k * 128 + n] : (_Float16)0.0f;
  }
  return b;
}

// ---------------------------------------------------------------- utility kernels
__global__ void k_zero(float* p, int n) {
  const int i = blockIdx.x * 256 + threadIdx.x;
  if (i < n) p[i] = 0.0f;
}

__global__ void k_f2h(const float* __restrict__ s, _Float16* __restrict__ d, int n) {
  const int i = blockIdx.x * 256 + threadIdx.x;
  if (i < n) d[i] = (_Float16)s[i];
}

// per-column sum / sumsq of x [kN][48]
__global__ void k_instats(const float* __restrict__ x, float* __restrict__ gsum,
                          float* __restrict__ gss) {
  const int t = threadIdx.x;
  if (t >= 240) return;
  const int col = t % 48, r0 = t / 48;
  const int base = blockIdx.x * 512;
  const int rend = (base + 512 < kN) ? base + 512 : kN;
  float s1 = 0.f, s2 = 0.f;
  for (int r = base + r0; r < rend; r += 5) {
    const float v = x[r * 48 + col];
    s1 += v; s2 += v * v;
  }
  unsafeAtomicAdd(&gsum[col], s1);
  unsafeAtomicAdd(&gss[col], s2);
}

// BN coefficients: A = g*rsqrt(var+eps), C = be - mean*A
__global__ void k_bncoef(const float* __restrict__ gsum, const float* __restrict__ gss,
                         float count, const float* __restrict__ g,
                         const float* __restrict__ be, float* __restrict__ A,
                         float* __restrict__ C, int n) {
  const int t = threadIdx.x;
  if (t < n) {
    const float m = gsum[t] / count;
    float v = gss[t] / count - m * m;
    v = v < 0.f ? 0.f : v;
    const float a = g[t] * rsqrtf(v + kEPS);
    A[t] = a;
    C[t] = be[t] - m * a;
  }
}

__global__ void k_applyin(const float* __restrict__ x, const float* __restrict__ A,
                          const float* __restrict__ C, _Float16* __restrict__ xh,
                          int n) {
  const int i = blockIdx.x * 256 + threadIdx.x;
  if (i < n) {
    const int c = i % 48;
    xh[i] = (_Float16)(x[i] * A[c] + C[c]);
  }
}

// ---------------------------------------------------------------- edge GEMM1
__global__ __launch_bounds__(256) void k_edge1(
    const _Float16* __restrict__ xh, const int* __restrict__ ei,
    const _Float16* __restrict__ w1, const float* __restrict__ b1,
    _Float16* __restrict__ Z, float* __restrict__ gsum, float* __restrict__ gss,
    int nE) {
  __shared__ __align__(16) _Float16 sA[16 * 96];
  __shared__ float sStat[256];
  const int t = threadIdx.x, lane = t & 31, w = t >> 5;
  const int n0 = w << 4, nloc = lane & 15, mb = (lane >> 4) << 3;
  sStat[t] = 0.f;
  v16h B0 = frag_b(w1, 96, n0, 0, lane);
  v16h B1 = frag_b(w1, 96, n0, 32, lane);
  v16h B2 = frag_b(w1, 96, n0, 64, lane);
  const float bias = b1[n0 + nloc];
  const int tiles = nE / 16;
  for (int tt = 0; tt < TPB; ++tt) {
    const int t0 = blockIdx.x * TPB + tt;
    if (t0 >= tiles) break;
    __syncthreads();
    if (t < 192) {  // gather 16 rows x (48 | 48) halves as 16B quads
      const int r = t / 12, q = t % 12;
      const int e = t0 * 16 + r;
      const int node = (q < 6) ? ei[e] : ei[nE + e];
      const int cq = (q < 6) ? q : q - 6;
      ((uint4*)sA)[r * 12 + q] = ((const uint4*)xh)[(size_t)node * 6 + cq];
    }
    __syncthreads();
    v8f acc;
#pragma unroll
    for (int r = 0; r < 8; ++r) acc[r] = bias;
    acc = wmma16(frag_a(sA, 96, 0, lane), B0, acc);
    acc = wmma16(frag_a(sA, 96, 32, lane), B1, acc);
    acc = wmma16(frag_a(sA, 96, 64, lane), B2, acc);
    float s1 = 0.f, s2 = 0.f;
#pragma unroll
    for (int r = 0; r < 8; ++r) {
      const float v = acc[r];
      s1 += v; s2 += v * v;
      Z[(size_t)(t0 * 16 + mb + r) * 128 + n0 + nloc] = (_Float16)v;
    }
    atomicAdd(&sStat[n0 + nloc], s1);
    atomicAdd(&sStat[128 + n0 + nloc], s2);
  }
  __syncthreads();
  if (t < 128) {
    unsafeAtomicAdd(&gsum[t], sStat[t]);
    unsafeAtomicAdd(&gss[t], sStat[128 + t]);
  }
}

// ---------------------------------------------------------------- fused edge GEMM2 + node1 GEMM1
__global__ __launch_bounds__(256) void k_edge2_node1a(
    const _Float16* __restrict__ xh, const int* __restrict__ ei,
    const _Float16* __restrict__ w2e, const float* __restrict__ b2e,
    const _Float16* __restrict__ w1n, const float* __restrict__ b1n,
    const float* __restrict__ cA, const float* __restrict__ cC,
    _Float16* __restrict__ Z, float* __restrict__ gsum, float* __restrict__ gss,
    int nE) {
  __shared__ __align__(16) _Float16 sZ[16 * 128];
  __shared__ __align__(16) _Float16 sA[16 * 192];
  __shared__ float sC[256];
  __shared__ float sStat[256];
  const int t = threadIdx.x, lane = t & 31, w = t >> 5;
  const int n0 = w << 4, nloc = lane & 15, mb = (lane >> 4) << 3;
  sStat[t] = 0.f;
  if (t < 128) { sC[t] = cA[t]; sC[128 + t] = cC[t]; }
  v16h Be[4], Bn[6];
#pragma unroll
  for (int c = 0; c < 4; ++c) Be[c] = frag_b(w2e, 128, n0, 32 * c, lane);
#pragma unroll
  for (int c = 0; c < 6; ++c) Bn[c] = frag_b(w1n, 176, n0, 32 * c, lane);
  const float biasE = b2e[n0 + nloc], biasN = b1n[n0 + nloc];
  const int tiles = nE / 16;
  for (int tt = 0; tt < TPB; ++tt) {
    const int t0 = blockIdx.x * TPB + tt;
    if (t0 >= tiles) break;
    __syncthreads();
    stage16B_issue(Z + (size_t)t0 * 2048, sZ, t);           // Z1 tile (async)
    if (t0 + 1 < tiles)
      __builtin_prefetch(Z + (size_t)(t0 + 1) * 2048 + t * 8, 0, 1);
    if (t < 96) {  // xh[row] into cols 0..47
      const int r = t / 6, q = t % 6;
      const int node = ei[t0 * 16 + r];
      ((uint4*)sA)[r * 24 + q] = ((const uint4*)xh)[(size_t)node * 6 + q];
    } else if (t < 128) {  // zero pad cols 176..191
      const int r = (t - 96) >> 1, j = (t - 96) & 1;
      ((uint4*)sA)[r * 24 + 22 + j] = make_uint4(0, 0, 0, 0);
    }
    stage_wait();
    __syncthreads();
#pragma unroll
    for (int j = 0; j < 8; ++j) {  // BN + ReLU in place on Z1 tile
      const int idx = t * 8 + j, c = idx & 127;
      float z = (float)sZ[idx];
      z = z * sC[c] + sC[128 + c];
      sZ[idx] = (_Float16)(z > 0.f ? z : 0.f);
    }
    __syncthreads();
    v8f acc;
#pragma unroll
    for (int r = 0; r < 8; ++r) acc[r] = biasE;
#pragma unroll
    for (int c = 0; c < 4; ++c)
      acc = wmma16(frag_a(sZ, 128, 32 * c, lane), Be[c], acc);
#pragma unroll
    for (int r = 0; r < 8; ++r)  // edge_attr into cols 48..175
      sA[(mb + r) * 192 + 48 + n0 + nloc] = (_Float16)acc[r];
    __syncthreads();
    v8f acc2;
#pragma unroll
    for (int r = 0; r < 8; ++r) acc2[r] = biasN;
#pragma unroll
    for (int c = 0; c < 6; ++c)
      acc2 = wmma16(frag_a(sA, 192, 32 * c, lane), Bn[c], acc2);
    float s1 = 0.f, s2 = 0.f;
#pragma unroll
    for (int r = 0; r < 8; ++r) {
      const float v = acc2[r];
      s1 += v; s2 += v * v;
      Z[(size_t)(t0 * 16 + mb + r) * 128 + n0 + nloc] = (_Float16)v;  // Z2 in place
    }
    atomicAdd(&sStat[n0 + nloc], s1);
    atomicAdd(&sStat[128 + n0 + nloc], s2);
  }
  __syncthreads();
  if (t < 128) {
    unsafeAtomicAdd(&gsum[t], sStat[t]);
    unsafeAtomicAdd(&gss[t], sStat[128 + t]);
  }
}

// ---------------------------------------------------------------- node1 GEMM2 + scatter-add
__global__ __launch_bounds__(256) void k_node1b(
    const int* __restrict__ ei, const _Float16* __restrict__ w2,
    const float* __restrict__ b2, const float* __restrict__ cA,
    const float* __restrict__ cC, const _Float16* __restrict__ Z,
    float* __restrict__ aggsum, float* __restrict__ cnt, int nE) {
  __shared__ __align__(16) _Float16 sZ[16 * 128];
  __shared__ float sC[256];
  const int t = threadIdx.x, lane = t & 31, w = t >> 5;
  const int n0 = w << 4, nloc = lane & 15, mb = (lane >> 4) << 3;
  if (t < 128) { sC[t] = cA[t]; sC[128 + t] = cC[t]; }
  v16h B[4];
#pragma unroll
  for (int c = 0; c < 4; ++c) B[c] = frag_b(w2, 128, n0, 32 * c, lane);
  const float bias = b2[n0 + nloc];
  const int tiles = nE / 16;
  for (int tt = 0; tt < TPB; ++tt) {
    const int t0 = blockIdx.x * TPB + tt;
    if (t0 >= tiles) break;
    __syncthreads();
    stage16B_issue(Z + (size_t)t0 * 2048, sZ, t);
    if (t0 + 1 < tiles)
      __builtin_prefetch(Z + (size_t)(t0 + 1) * 2048 + t * 8, 0, 1);
    if (t < 16) unsafeAtomicAdd(&cnt[ei[nE + t0 * 16 + t]], 1.0f);
    stage_wait();
    __syncthreads();
#pragma unroll
    for (int j = 0; j < 8; ++j) {
      const int idx = t * 8 + j, c = idx & 127;
      float z = (float)sZ[idx];
      z = z * sC[c] + sC[128 + c];
      sZ[idx] = (_Float16)(z > 0.f ? z : 0.f);
    }
    __syncthreads();
    v8f acc;
#pragma unroll
    for (int r = 0; r < 8; ++r) acc[r] = bias;
#pragma unroll
    for (int c = 0; c < 4; ++c)
      acc = wmma16(frag_a(sZ, 128, 32 * c, lane), B[c], acc);
#pragma unroll
    for (int r = 0; r < 8; ++r) {
      const int e = t0 * 16 + mb + r;
      const int cn = ei[nE + e];
      unsafeAtomicAdd(&aggsum[(size_t)cn * 128 + n0 + nloc], acc[r]);
    }
  }
}

// ---------------------------------------------------------------- node2 GEMM1
__global__ __launch_bounds__(256) void k_node2a(
    const _Float16* __restrict__ xh, const float* __restrict__ aggsum,
    const float* __restrict__ cnt, const _Float16* __restrict__ w1,
    const float* __restrict__ b1, _Float16* __restrict__ Z3,
    float* __restrict__ gsum, float* __restrict__ gss, int rows) {
  __shared__ __align__(16) _Float16 sA[16 * 192];
  __shared__ float sStat[256];
  const int t = threadIdx.x, lane = t & 31, w = t >> 5;
  const int n0 = w << 4, nloc = lane & 15, mb = (lane >> 4) << 3;
  sStat[t] = 0.f;
  v16h B[6];
#pragma unroll
  for (int c = 0; c < 6; ++c) B[c] = frag_b(w1, 176, n0, 32 * c, lane);
  const float bias = b1[n0 + nloc];
  const int tiles = rows / 16;
  for (int tt = 0; tt < TPB; ++tt) {
    const int t0 = blockIdx.x * TPB + tt;
    if (t0 >= tiles) break;
    __syncthreads();
    if (t0 + 1 < tiles)
      __builtin_prefetch(aggsum + (size_t)(t0 + 1) * 16 * 128 + t * 8, 0, 1);
    if (t < 96) {  // xh rows (contiguous) into cols 0..47
      const int r = t / 6, q = t % 6;
      ((uint4*)sA)[r * 24 + q] = ((const uint4*)xh)[(size_t)(t0 * 16 + r) * 6 + q];
    }
    {  // agg = aggsum / max(cnt,1) into cols 48..175
      const int r = t >> 4, c8 = (t & 15) << 3;
      const int node = t0 * 16 + r;
      const float rin = 1.0f / fmaxf(cnt[node], 1.0f);
#pragma unroll
      for (int j = 0; j < 8; ++j)
        sA[r * 192 + 48 + c8 + j] =
            (_Float16)(aggsum[(size_t)node * 128 + c8 + j] * rin);
    }
    if (t < 32) {  // zero pad cols 176..191
      const int r = t >> 1, j = t & 1;
      ((uint4*)sA)[r * 24 + 22 + j] = make_uint4(0, 0, 0, 0);
    }
    __syncthreads();
    v8f acc;
#pragma unroll
    for (int r = 0; r < 8; ++r) acc[r] = bias;
#pragma unroll
    for (int c = 0; c < 6; ++c)
      acc = wmma16(frag_a(sA, 192, 32 * c, lane), B[c], acc);
    float s1 = 0.f, s2 = 0.f;
#pragma unroll
    for (int r = 0; r < 8; ++r) {
      const float v = acc[r];
      s1 += v; s2 += v * v;
      Z3[(size_t)(t0 * 16 + mb + r) * 128 + n0 + nloc] = (_Float16)v;
    }
    atomicAdd(&sStat[n0 + nloc], s1);
    atomicAdd(&sStat[128 + n0 + nloc], s2);
  }
  __syncthreads();
  if (t < 128) {
    unsafeAtomicAdd(&gsum[t], sStat[t]);
    unsafeAtomicAdd(&gss[t], sStat[128 + t]);
  }
}

// ---------------------------------------------------------------- node2 GEMM2 + graph scatter
__global__ __launch_bounds__(256) void k_node2b(
    const int* __restrict__ batch, const _Float16* __restrict__ w2,
    const float* __restrict__ b2, const float* __restrict__ cA,
    const float* __restrict__ cC, const _Float16* __restrict__ Z3,
    float* __restrict__ usum, float* __restrict__ cntg, int rows) {
  __shared__ __align__(16) _Float16 sZ[16 * 128];
  __shared__ float sC[256];
  const int t = threadIdx.x, lane = t & 31, w = t >> 5;
  const int n0 = w << 4, nloc = lane & 15, mb = (lane >> 4) << 3;
  if (t < 128) { sC[t] = cA[t]; sC[128 + t] = cC[t]; }
  v16h B[4];
#pragma unroll
  for (int c = 0; c < 4; ++c) B[c] = frag_b(w2, 128, n0, 32 * c, lane);
  const float bias = b2[n0 + nloc];
  const int tiles = rows / 16;
  for (int tt = 0; tt < TPB; ++tt) {
    const int t0 = blockIdx.x * TPB + tt;
    if (t0 >= tiles) break;
    __syncthreads();
    stage16B_issue(Z3 + (size_t)t0 * 2048, sZ, t);
    if (t0 + 1 < tiles)
      __builtin_prefetch(Z3 + (size_t)(t0 + 1) * 2048 + t * 8, 0, 1);
    if (t < 16) unsafeAtomicAdd(&cntg[batch[t0 * 16 + t]], 1.0f);
    stage_wait();
    __syncthreads();
#pragma unroll
    for (int j = 0; j < 8; ++j) {
      const int idx = t * 8 + j, c = idx & 127;
      float z = (float)sZ[idx];
      z = z * sC[c] + sC[128 + c];
      sZ[idx] = (_Float16)(z > 0.f ? z : 0.f);
    }
    __syncthreads();
    v8f acc;
#pragma unroll
    for (int r = 0; r < 8; ++r) acc[r] = bias;
#pragma unroll
    for (int c = 0; c < 4; ++c)
      acc = wmma16(frag_a(sZ, 128, 32 * c, lane), B[c], acc);
#pragma unroll
    for (int r = 0; r < 8; ++r) {
      const int node = t0 * 16 + mb + r;
      unsafeAtomicAdd(&usum[(size_t)batch[node] * 128 + n0 + nloc], acc[r]);
    }
  }
}

// ---------------------------------------------------------------- global block (1 block, G=256 rows)
__global__ __launch_bounds__(256) void k_global(
    const float* __restrict__ usum, const float* __restrict__ cntg,
    const float* __restrict__ W1, const float* __restrict__ b1,
    const float* __restrict__ g, const float* __restrict__ be,
    const float* __restrict__ W2, const float* __restrict__ b2,
    float* __restrict__ zg, float* __restrict__ out) {
  __shared__ float sC[256];
  const int t = threadIdx.x;
  const float rin = 1.0f / fmaxf(cntg[t], 1.0f);
  for (int n = 0; n < 128; ++n) {
    float s = 0.f;
    for (int k = 0; k < 128; ++k) s += usum[t * 128 + k] * W1[k * 128 + n];
    zg[t * 128 + n] = s * rin + b1[n];
  }
  __threadfence_block();
  __syncthreads();
  if (t < 128) {  // per-column mean/var over 256 rows (no atomics)
    float s1 = 0.f, s2 = 0.f;
    for (int r = 0; r < kG; ++r) {
      const float v = zg[r * 128 + t];
      s1 += v; s2 += v * v;
    }
    const float m = s1 / (float)kG;
    float v = s2 / (float)kG - m * m;
    v = v < 0.f ? 0.f : v;
    const float a = g[t] * rsqrtf(v + kEPS);
    sC[t] = a;
    sC[128 + t] = be[t] - m * a;
  }
  __syncthreads();
  float o0 = b2[0], o1 = b2[1];
  for (int k = 0; k < 128; ++k) {
    float h = zg[t * 128 + k] * sC[k] + sC[128 + k];
    h = h > 0.f ? h : 0.f;
    o0 += h * W2[k * 2 + 0];
    o1 += h * W2[k * 2 + 1];
  }
  out[t * 2 + 0] = o0;
  out[t * 2 + 1] = o1;
}

// ---------------------------------------------------------------- host
extern "C" void kernel_launch(void* const* d_in, const int* in_sizes, int n_in,
                              void* d_out, int out_size, void* d_ws, size_t ws_size,
                              hipStream_t stream) {
  (void)in_sizes; (void)n_in; (void)out_size; (void)ws_size;
  const float* x    = (const float*)d_in[0];
  const int*   ei   = (const int*)d_in[1];
  const int*   batch= (const int*)d_in[2];
  const float* bn_g = (const float*)d_in[3];
  const float* bn_b = (const float*)d_in[4];
  const float* eW1  = (const float*)d_in[5];
  const float* eb1  = (const float*)d_in[6];
  const float* eg   = (const float*)d_in[7];
  const float* ebe  = (const float*)d_in[8];
  const float* eW2  = (const float*)d_in[9];
  const float* eb2  = (const float*)d_in[10];
  const float* n1W1 = (const float*)d_in[11];
  const float* n1b1 = (const float*)d_in[12];
  const float* n1g  = (const float*)d_in[13];
  const float* n1be = (const float*)d_in[14];
  const float* n1W2 = (const float*)d_in[15];
  const float* n1b2 = (const float*)d_in[16];
  const float* n2W1 = (const float*)d_in[17];
  const float* n2b1 = (const float*)d_in[18];
  const float* n2g  = (const float*)d_in[19];
  const float* n2be = (const float*)d_in[20];
  const float* n2W2 = (const float*)d_in[21];
  const float* n2b2 = (const float*)d_in[22];
  const float* glW1 = (const float*)d_in[23];
  const float* glb1 = (const float*)d_in[24];
  const float* glg  = (const float*)d_in[25];
  const float* glbe = (const float*)d_in[26];
  const float* glW2 = (const float*)d_in[27];
  const float* glb2 = (const float*)d_in[28];
  float* out = (float*)d_out;

  char* ws = (char*)d_ws;
  size_t off = 0;
  auto take = [&](size_t bytes) -> void* {
    void* p = ws + off;
    off = (off + bytes + 255) & ~(size_t)255;
    return p;
  };
  // ---- zero-initialized region (one contiguous span) ----
  float* aggsum = (float*)take((size_t)kN * 128 * 4);
  float* cnt    = (float*)take((size_t)kN * 4);
  float* usum   = (float*)take((size_t)kG * 128 * 4);
  float* cntg   = (float*)take((size_t)kG * 4);
  float* stats  = (float*)take(864 * 4);  // 2*48 + 6*128
  const size_t zeroBytes = off;
  // ---- non-zeroed scratch ----
  float* coefInA = (float*)take(48 * 4);
  float* coefInC = (float*)take(48 * 4);
  float* coefEA  = (float*)take(128 * 4);
  float* coefEC  = (float*)take(128 * 4);
  float* coefN1A = (float*)take(128 * 4);
  float* coefN1C = (float*)take(128 * 4);
  float* coefN2A = (float*)take(128 * 4);
  float* coefN2C = (float*)take(128 * 4);
  float* zg      = (float*)take((size_t)kG * 128 * 4);
  _Float16* xh   = (_Float16*)take((size_t)kN * 48 * 2);
  _Float16* ew1h = (_Float16*)take(96 * 128 * 2);
  _Float16* ew2h = (_Float16*)take(128 * 128 * 2);
  _Float16* m1w1h= (_Float16*)take(176 * 128 * 2);
  _Float16* m1w2h= (_Float16*)take(128 * 128 * 2);
  _Float16* m2w1h= (_Float16*)take(176 * 128 * 2);
  _Float16* m2w2h= (_Float16*)take(128 * 128 * 2);
  _Float16* Z    = (_Float16*)take((size_t)kE * 128 * 2);

  float* in_sum = stats;       float* in_ss = stats + 48;
  float* e_sum  = stats + 96;  float* e_ss  = stats + 224;
  float* m1_sum = stats + 352; float* m1_ss = stats + 480;
  float* m2_sum = stats + 608; float* m2_ss = stats + 736;

  const int zeroN = (int)(zeroBytes / 4);
  k_zero<<<(zeroN + 255) / 256, 256, 0, stream>>>((float*)ws, zeroN);

  k_f2h<<<(96 * 128 + 255) / 256, 256, 0, stream>>>(eW1, ew1h, 96 * 128);
  k_f2h<<<(128 * 128 + 255) / 256, 256, 0, stream>>>(eW2, ew2h, 128 * 128);
  k_f2h<<<(176 * 128 + 255) / 256, 256, 0, stream>>>(n1W1, m1w1h, 176 * 128);
  k_f2h<<<(128 * 128 + 255) / 256, 256, 0, stream>>>(n1W2, m1w2h, 128 * 128);
  k_f2h<<<(176 * 128 + 255) / 256, 256, 0, stream>>>(n2W1, m2w1h, 176 * 128);
  k_f2h<<<(128 * 128 + 255) / 256, 256, 0, stream>>>(n2W2, m2w2h, 128 * 128);

  k_instats<<<(kN + 511) / 512, 256, 0, stream>>>(x, in_sum, in_ss);
  k_bncoef<<<1, 128, 0, stream>>>(in_sum, in_ss, (float)kN, bn_g, bn_b,
                                  coefInA, coefInC, 48);
  k_applyin<<<(kN * 48 + 255) / 256, 256, 0, stream>>>(x, coefInA, coefInC, xh,
                                                       kN * 48);

  const int TEb = (kE / 16 + TPB - 1) / TPB;
  const int TNb = (kN / 16 + TPB - 1) / TPB;

  k_edge1<<<TEb, 256, 0, stream>>>(xh, ei, ew1h, eb1, Z, e_sum, e_ss, kE);
  k_bncoef<<<1, 128, 0, stream>>>(e_sum, e_ss, (float)kE, eg, ebe, coefEA, coefEC, 128);
  k_edge2_node1a<<<TEb, 256, 0, stream>>>(xh, ei, ew2h, eb2, m1w1h, n1b1,
                                          coefEA, coefEC, Z, m1_sum, m1_ss, kE);
  k_bncoef<<<1, 128, 0, stream>>>(m1_sum, m1_ss, (float)kE, n1g, n1be,
                                  coefN1A, coefN1C, 128);
  k_node1b<<<TEb, 256, 0, stream>>>(ei, m1w2h, n1b2, coefN1A, coefN1C, Z,
                                    aggsum, cnt, kE);
  k_node2a<<<TNb, 256, 0, stream>>>(xh, aggsum, cnt, m2w1h, n2b1, Z,
                                    m2_sum, m2_ss, kN);
  k_bncoef<<<1, 128, 0, stream>>>(m2_sum, m2_ss, (float)kN, n2g, n2be,
                                  coefN2A, coefN2C, 128);
  k_node2b<<<TNb, 256, 0, stream>>>(batch, m2w2h, n2b2, coefN2A, coefN2C, Z,
                                    usum, cntg, kN);
  k_global<<<1, 256, 0, stream>>>(usum, cntg, glW1, glb1, glg, glbe, glW2, glb2,
                                  zg, out);
}
